// AttentionalStackGCN_11424613008073
// MI455X (gfx1250) — compile-verified, matrix operand order checked by reading, not run on hardware
//
#include <hip/hip_runtime.h>

#define NU 3000
#define NV 4000
#define FIN 512
#define DOUT 500
#define SSUP 5
#define DS 100
#define NEGV -10000000000.0f
#define ALPHAV 0.2f

typedef __attribute__((ext_vector_type(2)))  float          v2f;
typedef __attribute__((ext_vector_type(4)))  float          v4f;
typedef __attribute__((ext_vector_type(8)))  float          v8f;
typedef __attribute__((ext_vector_type(16))) __bf16         v16bf;
typedef __attribute__((ext_vector_type(8)))  unsigned int   v8ui;
typedef __attribute__((ext_vector_type(4)))  unsigned int   v4ui;
typedef __attribute__((ext_vector_type(8)))  int            v8i;
typedef __attribute__((ext_vector_type(4)))  int            v4i;

static __device__ inline unsigned int f2bf_bits(float f) {
    unsigned int u = __float_as_uint(f);
    u += 0x7fffu + ((u >> 16) & 1u);   // round-to-nearest-even
    return u >> 16;
}

// ---------------------------------------------------------------------------
// Phase 1: Tmp = X (N x 512) @ W (512 x 500), fp32 WMMA 16x16x4.
// Block = 128 threads (4 waves) sharing one 16-row strip of X.
// The 16x64 f32 A tile is staged into LDS by the Tensor Data Mover:
//   - tile_dim0=64, tile_dim1=16, tensor_dim1=N-row0 (edge rows zero-filled)
//   - pad_interval=64 dwords, pad_amount=4 dwords -> 68-float LDS rows
//     (conflict-free ds reads: 68*r mod 64 banks all distinct)
// Wave 0 issues the TDM descriptor, waits TENSORcnt, then the block barriers.
// ---------------------------------------------------------------------------
__global__ void gemm_xw_kernel(const float* __restrict__ X,
                               const float* __restrict__ W,
                               float* __restrict__ Tmp, int N) {
    __shared__ float sA[16 * 68];                  // 16 x 64 tile, row pad 68

    const int tid  = threadIdx.x;
    const int wv   = tid >> 5;
    const int lane = tid & 31;
    const int rr   = lane & 15;
    const int hl   = lane >> 4;                    // half-wave: K pair select

    const int row0 = blockIdx.x * 16;
    const int col0 = (blockIdx.y * 4 + wv) * 16;   // 0..496

    const int  colb  = col0 + rr;
    const bool cok   = (colb < DOUT);
    const int  colbC = min(colb, DOUT - 1);

    v8f c = {};

    for (int kb = 0; kb < FIN; kb += 64) {
#if __has_builtin(__builtin_amdgcn_tensor_load_to_lds)
        if (tid < 32) {                            // wave 0 drives the TDM
            const unsigned long long gaddr =
                (unsigned long long)(uintptr_t)X +
                (((unsigned long long)row0 * FIN + kb) << 2);
            const unsigned int ldsOff = (unsigned int)(uintptr_t)(&sA[0]);

            // D# group 0: count=1 | lds_addr | global_addr[56:0] | type=2
            v4ui g0;
            g0[0] = 1u;                                        // count=1
            g0[1] = ldsOff;                                    // lds_addr
            g0[2] = (unsigned int)(gaddr & 0xffffffffu);       // gaddr[31:0]
            g0[3] = (unsigned int)((gaddr >> 32) & 0x1ffffffu) // gaddr[56:32]
                    | (2u << 30);                              // type=2
            // D# group 1: data_size=4B, pad 4 dw per 64 dw, dims/strides
            const unsigned int td0 = (unsigned int)(FIN - kb); // never OOB in x
            const unsigned int td1 = (unsigned int)(N - row0); // row clip -> 0
            v8i g1;
            g1[0] = (int)((2u << 16)        // data_size = 4 bytes
                        | (1u << 20)        // pad_enable
                        | (5u << 22)        // pad_interval: 64 dwords
                        | (3u << 25));      // pad_amount: 4 dwords
            g1[1] = (int)((td0 & 0xffffu) << 16);              // dim0[15:0]
            g1[2] = (int)(((td0 >> 16) & 0xffffu)
                        | ((td1 & 0xffffu) << 16));            // dim0hi|dim1lo
            g1[3] = (int)(((td1 >> 16) & 0xffffu)
                        | (64u << 16));                        // dim1hi|tile0=64
            g1[4] = 16;                                        // tile_dim1=16
            g1[5] = FIN;                                       // dim0_stride lo
            g1[6] = 0;
            g1[7] = 0;
            v4i gz4 = {0, 0, 0, 0};
            v8i gz8 = {0, 0, 0, 0, 0, 0, 0, 0};
            __builtin_amdgcn_tensor_load_to_lds(g0, g1, gz4, gz4, gz8, 0);
            __builtin_amdgcn_s_wait_tensorcnt(0);
        }
#else
        #pragma unroll
        for (int e = tid; e < 16 * 16; e += 128) { // 256 float4 elements
            const int r  = e >> 4;
            const int c4 = (e & 15) << 2;
            const int rowL = min(row0 + r, N - 1);
            const v4f xv = *(const v4f*)(X + (long)rowL * FIN + kb + c4);
            sA[r * 68 + c4 + 0] = xv.x;
            sA[r * 68 + c4 + 1] = xv.y;
            sA[r * 68 + c4 + 2] = xv.z;
            sA[r * 68 + c4 + 3] = xv.w;
        }
#endif
        __syncthreads();

        #pragma unroll 4
        for (int k = 0; k < 64; k += 4) {
            const int ka = k + 2 * hl;
            // A fragment from LDS (conflict-free: 68*rr banks distinct)
            v2f a = *(const v2f*)(sA + rr * 68 + ka);
            // B fragment: unconditional clamped loads + select
            const int kg = kb + ka;
            float bx = W[(long)kg       * DOUT + colbC];
            float by = W[(long)(kg + 1) * DOUT + colbC];
            v2f b;
            b.x = cok ? bx : 0.0f;
            b.y = cok ? by : 0.0f;
            c = __builtin_amdgcn_wmma_f32_16x16x4_f32(false, a, false, b,
                                                      (short)0, c, false, false);
        }
        __syncthreads();
    }

    #pragma unroll
    for (int i = 0; i < 8; ++i) {
        const int row = row0 + i + 8 * hl;
        if (row < N && cok) Tmp[(long)row * DOUT + colb] = c[i];
    }
}

// ---------------------------------------------------------------------------
// Phase 2a: attn[s*N+n] = sum_d Tmp[n][s*100+d] * avec[s*100+d]
// One wave per (s,n) row.
// ---------------------------------------------------------------------------
__global__ void attn_kernel(const float* __restrict__ Tmp,
                            const float* __restrict__ avec,
                            float* __restrict__ attn, int N) {
    const int lane = threadIdx.x & 31;
    const int wid  = (blockIdx.x * blockDim.x + threadIdx.x) >> 5;
    if (wid >= SSUP * N) return;
    const int s = wid / N;
    const int n = wid % N;

    float sum = 0.0f;
    for (int d = lane; d < DS; d += 32)
        sum += Tmp[(long)n * DOUT + s * DS + d] * avec[s * DS + d];
    #pragma unroll
    for (int off = 16; off > 0; off >>= 1)
        sum += __shfl_xor(sum, off, 32);
    if (lane == 0) attn[wid] = sum;
}

// ---------------------------------------------------------------------------
// Phase 2b: one-pass online softmax stats per row (one wave32 per row).
// Streams the 240MB mask exactly once; stores rowmax and 1/rowsum.
// ---------------------------------------------------------------------------
__global__ void softmax_stats_kernel(const int*   __restrict__ supp,
                                     const float* __restrict__ attnR,
                                     const float* __restrict__ attnC,
                                     float* __restrict__ rowmax,
                                     float* __restrict__ rowrinv,
                                     int Nrow, int Kdim) {
    const int lane = threadIdx.x & 31;
    const int wid  = (blockIdx.x * blockDim.x + threadIdx.x) >> 5;
    if (wid >= SSUP * Nrow) return;
    const int s = wid / Nrow;
    const int r = wid % Nrow;

    const float ar = attnR[s * Nrow + r];
    const int* srow = supp + ((long)s * Nrow + r) * Kdim;
    const float* ac = attnC + s * Kdim;

    float m = -3.0e38f, sm = 0.0f;
    for (int j = lane; j < Kdim; j += 32) {
        __builtin_prefetch(srow + j + 512, 0, 1);     // global_prefetch_b8
        float t = ar + ac[j];
        t = t > 0.0f ? t : ALPHAV * t;
        t = (srow[j] == 0) ? t + NEGV : t;
        const float M = fmaxf(m, t);
        sm = sm * __expf(m - M) + __expf(t - M);
        m = M;
    }
    #pragma unroll
    for (int off = 16; off > 0; off >>= 1) {
        const float m2 = __shfl_xor(m, off, 32);
        const float s2 = __shfl_xor(sm, off, 32);
        const float M  = fmaxf(m, m2);
        sm = sm * __expf(m - M) + s2 * __expf(m2 - M);
        m = M;
    }
    if (lane == 0) { rowmax[wid] = m; rowrinv[wid] = 1.0f / sm; }
}

// ---------------------------------------------------------------------------
// Phase 3: Z[r, s*100+c] = relu( sum_m P[s,r,m] * B[m, s*100+c] )
// P recomputed on the fly (one exp per mask element), staged bf16 in LDS
// (rows padded to 36 -> conflict-free ds_load_b32 A fragments), consumed by
// V_WMMA_F32_16X16X32_BF16. Block = 7 waves: 16 rows x 7 col tiles (112>=100).
// ---------------------------------------------------------------------------
__global__ void agg_gemm_kernel(const int*   __restrict__ supp,   // [S][Nrow][Kdim]
                                const float* __restrict__ attnR,  // [S][Nrow]
                                const float* __restrict__ attnC,  // [S][Kdim]
                                const float* __restrict__ rowmax,
                                const float* __restrict__ rowrinv,
                                const float* __restrict__ Bmat,   // [Kdim][500]
                                float* __restrict__ outp,         // [Nrow][500]
                                int Nrow, int Kdim) {
    __shared__ unsigned short pTile[16 * 36];

    const int tid  = threadIdx.x;
    const int wv   = tid >> 5;
    const int lane = tid & 31;
    const int rr   = lane & 15;
    const int hl   = lane >> 4;

    const int s    = blockIdx.y;
    const int row0 = blockIdx.x * 16;
    const int jcol  = wv * 16 + rr;              // 0..111 local output column
    const bool cok  = (jcol < DS);
    const long bcol = (long)s * DS + min(jcol, DS - 1);

    v8f c = {};

    for (int m0 = 0; m0 < Kdim; m0 += 32) {
        // --- cooperative bf16 probability tile (16 rows x 32 K) ------------
        #pragma unroll
        for (int e = tid; e < 16 * 32; e += 224) {
            const int pr = e >> 5, pc = e & 31;
            const int  row = row0 + pr, mm = m0 + pc;
            const bool ok  = (row < Nrow) & (mm < Kdim);
            const int  rowC = min(row, Nrow - 1);
            const int  mmC  = min(mm,  Kdim - 1);
            const int  sv   = supp[((long)s * Nrow + rowC) * Kdim + mmC];
            float t = attnR[s * Nrow + rowC] + attnC[s * Kdim + mmC];
            t = t > 0.0f ? t : ALPHAV * t;
            t = (sv == 0) ? t + NEGV : t;
            float p = __expf(t - rowmax[s * Nrow + rowC]) * rowrinv[s * Nrow + rowC];
            p = ok ? p : 0.0f;
            pTile[pr * 36 + pc] = (unsigned short)f2bf_bits(p);
        }
        __syncthreads();

        // --- A fragment from LDS (conflict-free ds_load_b32) ---------------
        const unsigned int* p32 =
            (const unsigned int*)(pTile + rr * 36 + hl * 16);
        v8ui aw;
        #pragma unroll
        for (int j = 0; j < 8; ++j) aw[j] = p32[j];

        // --- B fragment from global (L2-resident tmp), clamp+select --------
        v8ui bw;
        #pragma unroll
        for (int j = 0; j < 8; ++j) {
            const int k0 = 2 * j + 16 * hl;
            const int m1 = m0 + k0, m2 = m1 + 1;
            const float r0 = Bmat[(long)min(m1, Kdim - 1) * DOUT + bcol];
            const float r1 = Bmat[(long)min(m2, Kdim - 1) * DOUT + bcol];
            const float f0 = (cok && m1 < Kdim) ? r0 : 0.0f;
            const float f1 = (cok && m2 < Kdim) ? r1 : 0.0f;
            bw[j] = f2bf_bits(f0) | (f2bf_bits(f1) << 16);
        }

        c = __builtin_amdgcn_wmma_f32_16x16x32_bf16(
                false, __builtin_bit_cast(v16bf, aw),
                false, __builtin_bit_cast(v16bf, bw),
                (short)0, c, false, false);
        __syncthreads();
    }

    #pragma unroll
    for (int i = 0; i < 8; ++i) {
        const int row = row0 + i + 8 * hl;
        if (row < Nrow && cok)
            outp[(long)row * DOUT + s * DS + jcol] = fmaxf(c[i], 0.0f);
    }
}

// ---------------------------------------------------------------------------
extern "C" void kernel_launch(void* const* d_in, const int* in_sizes, int n_in,
                              void* d_out, int out_size, void* d_ws, size_t ws_size,
                              hipStream_t stream) {
    const float* x_u   = (const float*)d_in[0];
    const float* x_v   = (const float*)d_in[1];
    const float* W     = (const float*)d_in[2];
    const float* a_u   = (const float*)d_in[3];
    const float* a_v   = (const float*)d_in[4];
    const int*   supp  = (const int*)d_in[5];
    const int*   suppT = (const int*)d_in[6];

    float* ws     = (float*)d_ws;
    float* tmp_u  = ws;                       // 3000*500
    float* tmp_v  = ws + 1500000;             // 4000*500
    float* attn_u = ws + 3500000;             // 5*3000
    float* attn_v = ws + 3515000;             // 5*4000
    float* max_u  = ws + 3535000;
    float* rinv_u = ws + 3550000;
    float* max_v  = ws + 3565000;
    float* rinv_v = ws + 3585000;

    float* z_u = (float*)d_out;               // 3000*500
    float* z_v = z_u + (long)NU * DOUT;       // 4000*500

    // Phase 1: fp32 WMMA feature GEMMs (TDM-staged A tiles)
    hipLaunchKernelGGL(gemm_xw_kernel, dim3((NU + 15) / 16, 8), dim3(128), 0,
                       stream, x_u, W, tmp_u, NU);
    hipLaunchKernelGGL(gemm_xw_kernel, dim3((NV + 15) / 16, 8), dim3(128), 0,
                       stream, x_v, W, tmp_v, NV);

    // Phase 2a: attention logit vectors
    hipLaunchKernelGGL(attn_kernel, dim3((SSUP * NU + 7) / 8), dim3(256), 0,
                       stream, tmp_u, a_u, attn_u, NU);
    hipLaunchKernelGGL(attn_kernel, dim3((SSUP * NV + 7) / 8), dim3(256), 0,
                       stream, tmp_v, a_v, attn_v, NV);

    // Phase 2b: single-pass online softmax stats over the mask stream
    hipLaunchKernelGGL(softmax_stats_kernel, dim3((SSUP * NU + 7) / 8), dim3(256),
                       0, stream, supp, attn_u, attn_v, max_u, rinv_u, NU, NV);
    hipLaunchKernelGGL(softmax_stats_kernel, dim3((SSUP * NV + 7) / 8), dim3(256),
                       0, stream, suppT, attn_v, attn_u, max_v, rinv_v, NV, NU);

    // Phase 3: bf16 WMMA aggregation GEMMs with fused on-the-fly softmax P
    hipLaunchKernelGGL(agg_gemm_kernel, dim3((NU + 15) / 16, SSUP), dim3(224), 0,
                       stream, supp, attn_u, attn_v, max_u, rinv_u,
                       tmp_v, z_u, NU, NV);
    hipLaunchKernelGGL(agg_gemm_kernel, dim3((NV + 15) / 16, SSUP), dim3(224), 0,
                       stream, suppT, attn_v, attn_u, max_v, rinv_v,
                       tmp_u, z_v, NV, NU);
}
